// GATNet_72164040508105
// MI455X (gfx1250) — matrix operand order repaired and checked by reading, not compiled, not run on hardware
//
#include <hip/hip_runtime.h>
#include <hip/hip_bf16.h>
#include <math.h>

// ---------------- problem constants (match reference) ----------------
static constexpr int Nn   = 20000;     // nodes
static constexpr int Ee   = 320000;    // directed edges (before self loops)
static constexpr int EP   = Ee + Nn;   // edges incl. self loops
static constexpr int Gg   = 512;       // graphs
static constexpr int IN_CH= 3;
static constexpr int Hh   = 4;         // heads
static constexpr int Cc   = 64;        // channels per head
static constexpr int HC   = Hh * Cc;   // 256
static constexpr int FC   = 128;
static constexpr int BW   = 7;
static constexpr float NEG_SLOPE = 0.2f;

typedef __attribute__((ext_vector_type(16))) _Float16 v16h;
typedef __attribute__((ext_vector_type(8)))  float    v8f;

static inline int cdiv(long a, long b) { return (int)((a + b - 1) / b); }

// ---------------- small helpers ----------------
__global__ void fill_u32_kernel(unsigned int* __restrict__ p, unsigned int v, long n) {
    long i = (long)blockIdx.x * blockDim.x + threadIdx.x;
    if (i < n) p[i] = v;
}

// float atomic max via monotone int mapping (init with -inf bits 0xFF800000)
__device__ __forceinline__ void atomicMaxFloat(float* addr, float val) {
    int iv = __float_as_int(val);
    if (iv >= 0) atomicMax((int*)addr, iv);
    else         atomicMin((unsigned int*)addr, (unsigned int)iv);
}

__device__ __forceinline__ void edge_sd(const int* __restrict__ ei, int e, int& s, int& d) {
    if (e < Ee) { s = ei[e]; d = ei[Ee + e]; }
    else        { s = e - Ee; d = e - Ee; }      // self loop
}

// ---------------- layer-1 linear: h = x @ W1  (K=3, memory bound) ----------------
__global__ void lin1_kernel(const float* __restrict__ x, const float* __restrict__ W1,
                            float* __restrict__ h) {
    long t = (long)blockIdx.x * blockDim.x + threadIdx.x;
    if (t >= (long)Nn * HC) return;
    int n = (int)(t / HC), c = (int)(t % HC);
    float v = x[n*IN_CH+0]*W1[0*HC+c] + x[n*IN_CH+1]*W1[1*HC+c] + x[n*IN_CH+2]*W1[2*HC+c];
    h[t] = v;
}

// ---------------- attention coefficients a_s/a_d : [N,H] ----------------
__global__ void attcoef_kernel(const float* __restrict__ h,
                               const float* __restrict__ att_src,
                               const float* __restrict__ att_dst,
                               float* __restrict__ a_s, float* __restrict__ a_d) {
    int t = blockIdx.x * blockDim.x + threadIdx.x;
    if (t >= Nn * Hh) return;
    int n = t / Hh, hd = t % Hh;
    const float* hr = h + (long)n*HC + hd*Cc;
    const float* as = att_src + hd*Cc;
    const float* ad = att_dst + hd*Cc;
    float s = 0.f, d = 0.f;
    #pragma unroll 4
    for (int c = 0; c < Cc; ++c) { s += hr[c]*as[c]; d += hr[c]*ad[c]; }
    a_s[t] = s; a_d[t] = d;
}

// ---------------- per-edge leaky-relu score, segment max over dst ----------------
__global__ void edge_max_kernel(const int* __restrict__ ei,
                                const float* __restrict__ a_s, const float* __restrict__ a_d,
                                float* __restrict__ mmax) {
    int t = blockIdx.x * blockDim.x + threadIdx.x;
    if (t >= EP * Hh) return;
    int e = t / Hh, hd = t % Hh;
    int s, d; edge_sd(ei, e, s, d);
    float v = a_s[s*Hh + hd] + a_d[d*Hh + hd];
    v = (v > 0.f) ? v : NEG_SLOPE * v;
    atomicMaxFloat(&mmax[d*Hh + hd], v);
}

// ---------------- exp(e - m[dst]) and segment-sum denom ----------------
__global__ void edge_exp_kernel(const int* __restrict__ ei,
                                const float* __restrict__ a_s, const float* __restrict__ a_d,
                                const float* __restrict__ mmax,
                                float* __restrict__ exb, float* __restrict__ denom) {
    int t = blockIdx.x * blockDim.x + threadIdx.x;
    if (t >= EP * Hh) return;
    int e = t / Hh, hd = t % Hh;
    int s, d; edge_sd(ei, e, s, d);
    float v = a_s[s*Hh + hd] + a_d[d*Hh + hd];
    v = (v > 0.f) ? v : NEG_SLOPE * v;
    float ex = __expf(v - mmax[d*Hh + hd]);
    exb[t] = ex;
    atomicAdd(&denom[d*Hh + hd], ex);
}

// ---------------- weighted scatter: out[dst] += alpha * h[src] ----------------
// One wave32 per edge; lane owns 8 contiguous channels -> h[src] gather is
// two global_load_b128 per lane (512B per wave-instruction); 8 f32 L2 atomics.
__global__ void scatter_kernel(const int* __restrict__ ei,
                               const float* __restrict__ h,
                               const float* __restrict__ exb, const float* __restrict__ denom,
                               float* __restrict__ out) {
    long gtid = (long)blockIdx.x * blockDim.x + threadIdx.x;
    int e = (int)(gtid >> 5);          // one wave32 per edge
    int lane = (int)(gtid & 31);
    if (e >= EP) return;
    int s, d; edge_sd(ei, e, s, d);
    int c0 = lane * 8;                 // 32 lanes x 8 ch = 256
    int hd = lane >> 3;                // c0 / 64
    float alpha = exb[e*Hh + hd] / denom[d*Hh + hd];
    const float4* hs = (const float4*)(h + (long)s*HC + c0);
    float4 v0 = hs[0], v1 = hs[1];
    float* od = out + (long)d*HC + c0;
    atomicAdd(&od[0], alpha * v0.x);
    atomicAdd(&od[1], alpha * v0.y);
    atomicAdd(&od[2], alpha * v0.z);
    atomicAdd(&od[3], alpha * v0.w);
    atomicAdd(&od[4], alpha * v1.x);
    atomicAdd(&od[5], alpha * v1.y);
    atomicAdd(&od[6], alpha * v1.z);
    atomicAdd(&od[7], alpha * v1.w);
}

// ---------------- bias + relu; optional f16 copy for next WMMA GEMM ----------------
__global__ void bias_relu_kernel(float* __restrict__ h, const float* __restrict__ b,
                                 _Float16* __restrict__ h16) {
    long t = (long)blockIdx.x * blockDim.x + threadIdx.x;
    if (t >= (long)Nn * HC) return;
    int c = (int)(t % HC);
    float v = fmaxf(h[t] + b[c], 0.f);
    h[t] = v;
    if (h16) h16[t] = (_Float16)v;
}

// ---------------- transpose + f32->f16 convert: src[rows,cols] -> dstT[cols,rows] ----------------
__global__ void transpose_f16_kernel(const float* __restrict__ src, _Float16* __restrict__ dstT,
                                     int rows, int cols) {
    int t = blockIdx.x * blockDim.x + threadIdx.x;
    if (t >= rows * cols) return;
    int r = t / cols, c = t % cols;
    dstT[(long)c*rows + r] = (_Float16)src[t];
}

// ---------------- WMMA GEMM: C[M,N] = A[M,K](f16,row) @ Bt[N,K](f16,row)^T ----------------
// One wave per 16x64 output strip: the A fragment is loaded once per k-step and
// reused across 4 WMMAs (4 N-tiles), cutting A traffic 4x. Requires N % 64 == 0,
// M % 16 == 0, K % 32 == 0. Fragment layouts per CDNA5 ISA 7.12.2.
__global__ __launch_bounds__(256) void wmma_gemm_kernel(
    const _Float16* __restrict__ A, const _Float16* __restrict__ Bt,
    float* __restrict__ Cmat, const float* __restrict__ bias,
    int M, int N, int K, int do_relu)
{
    constexpr int NB = 4;                  // N-tiles per wave (16*NB = 64 cols)
    int wavesPerBlock = blockDim.x >> 5;
    long strip = (long)blockIdx.x * wavesPerBlock + (threadIdx.x >> 5);
    int lane  = threadIdx.x & 31;
    int nstrips = N / (16 * NB);
    int mt = (int)(strip / nstrips);
    int nb = (int)(strip % nstrips);
    if (mt >= (M >> 4)) return;            // wave-uniform exit: EXEC stays all-ones for WMMA

    int grp  = lane >> 4;                  // 0: lanes 0-15, 1: lanes 16-31
    int mrow = mt*16 + (lane & 15);        // A row owned by this lane
    int ncol = nb*(16*NB) + (lane & 15);   // first B column owned by this lane

    v8f acc[NB];
    #pragma unroll
    for (int t = 0; t < NB; ++t) acc[t] = (v8f){};

    for (int kk = 0; kk < K; kk += 32) {
        union frag { v16h v; unsigned int u[8]; };
        frag af, bf[NB];
        #pragma unroll
        for (int r = 0; r < 8; ++r) {
            // A 16x32 f16: VGPR r holds K pair; K = (r<4?0:16) + grp*8 + (r&3)*2
            int ka = ((r < 4) ? 0 : 16) + grp*8 + (r & 3)*2;
            af.u[r] = *(const unsigned int*)(A + (long)mrow*K + kk + ka);
        }
        #pragma unroll
        for (int t = 0; t < NB; ++t) {
            #pragma unroll
            for (int r = 0; r < 8; ++r) {
                // B 32x16 f16 (from Bt[N,K]): lane owns one column; K = (r<4?0:16) + grp*8 + (r&3)*2
                int kb = ((r < 4) ? 0 : 16) + grp*8 + (r & 3)*2;
                bf[t].u[r] = *(const unsigned int*)(Bt + (long)(ncol + t*16)*K + kk + kb);
            }
        }
        #pragma unroll
        for (int t = 0; t < NB; ++t) {
            acc[t] = __builtin_amdgcn_wmma_f32_16x16x32_f16(
                         false, af.v, false, bf[t].v, (short)0, acc[t], false, false);
        }
    }
    #pragma unroll
    for (int t = 0; t < NB; ++t) {
        int col = ncol + t*16;
        #pragma unroll
        for (int r = 0; r < 8; ++r) {      // C/D: VGPR r -> row M = r + grp*8
            int row = mt*16 + grp*8 + r;
            float v = acc[t][r];
            if (bias)    v += bias[col];
            if (do_relu) v = fmaxf(v, 0.f);
            Cmat[(long)row*N + col] = v;
        }
    }
}

// ---------------- global mean pool ----------------
__global__ void pool_sum_kernel(const float* __restrict__ h, const int* __restrict__ batch,
                                float* __restrict__ emb, float* __restrict__ cnt) {
    long t = (long)blockIdx.x * blockDim.x + threadIdx.x;
    if (t >= (long)Nn * HC) return;
    int n = (int)(t / HC), c = (int)(t % HC);
    int g = batch[n];
    atomicAdd(&emb[(long)g*HC + c], h[t]);
    if (c == 0) atomicAdd(&cnt[g], 1.0f);
}

__global__ void pool_div_kernel(const float* __restrict__ emb, const float* __restrict__ cnt,
                                _Float16* __restrict__ emb16) {
    int t = blockIdx.x * blockDim.x + threadIdx.x;
    if (t >= Gg * HC) return;
    int g = t / HC;
    emb16[t] = (_Float16)(emb[t] / fmaxf(cnt[g], 1.0f));
}

// ---------------- policy / value heads ----------------
__global__ void heads_kernel(const float* __restrict__ z,
                             const float* __restrict__ pol_w, const float* __restrict__ pol_b,
                             const float* __restrict__ val_w, const float* __restrict__ val_b,
                             float* __restrict__ out) {
    int g = blockIdx.x * blockDim.x + threadIdx.x;
    if (g >= Gg) return;
    const float* zr = z + (long)g*FC;
    #pragma unroll
    for (int j = 0; j < BW; ++j) {
        float s = pol_b[j];
        for (int k = 0; k < FC; ++k) s += zr[k] * pol_w[k*BW + j];
        out[g*BW + j] = s;
    }
    float s = val_b[0];
    for (int k = 0; k < FC; ++k) s += zr[k] * val_w[k];
    out[Gg*BW + g] = tanhf(s);
}

// ---------------- orchestration ----------------
extern "C" void kernel_launch(void* const* d_in, const int* in_sizes, int n_in,
                              void* d_out, int out_size, void* d_ws, size_t ws_size,
                              hipStream_t stream) {
    (void)in_sizes; (void)n_in; (void)out_size; (void)ws_size;

    const float* x        = (const float*)d_in[0];
    const int*   ei       = (const int*)  d_in[1];   // [2,E] flattened
    const int*   batch    = (const int*)  d_in[2];
    const float* W1       = (const float*)d_in[3];
    const float* att_s1   = (const float*)d_in[4];
    const float* att_d1   = (const float*)d_in[5];
    const float* b1       = (const float*)d_in[6];
    const float* W2       = (const float*)d_in[7];
    const float* att_s2   = (const float*)d_in[8];
    const float* att_d2   = (const float*)d_in[9];
    const float* b2       = (const float*)d_in[10];
    const float* fc1_w    = (const float*)d_in[11];
    const float* fc1_b    = (const float*)d_in[12];
    const float* pol_w    = (const float*)d_in[13];
    const float* pol_b    = (const float*)d_in[14];
    const float* val_w    = (const float*)d_in[15];
    const float* val_b    = (const float*)d_in[16];
    float* out = (float*)d_out;

    // -------- workspace carve-up --------
    char* base = (char*)d_ws; size_t off = 0;
    auto carve = [&](size_t bytes) -> void* {
        void* p = base + off; off += (bytes + 255) & ~(size_t)255; return p;
    };
    float*     P0    = (float*)    carve((size_t)Nn*HC*4);   // linear outputs
    float*     P1    = (float*)    carve((size_t)Nn*HC*4);   // aggregated outputs
    _Float16*  F0    = (_Float16*) carve((size_t)Nn*HC*2);   // f16 A for GEMM2
    float*     a_s   = (float*)    carve((size_t)Nn*Hh*4);
    float*     a_d   = (float*)    carve((size_t)Nn*Hh*4);
    float*     mmax  = (float*)    carve((size_t)Nn*Hh*4);
    float*     denom = (float*)    carve((size_t)Nn*Hh*4);
    float*     exb   = (float*)    carve((size_t)EP*Hh*4);
    _Float16*  W2t   = (_Float16*) carve((size_t)HC*HC*2);   // [N,K]
    _Float16*  fc1t  = (_Float16*) carve((size_t)FC*HC*2);   // [N,K]
    float*     embp  = (float*)    carve((size_t)Gg*HC*4);
    float*     cnt   = (float*)    carve((size_t)Gg*4);
    _Float16*  emb16 = (_Float16*) carve((size_t)Gg*HC*2);
    float*     z     = (float*)    carve((size_t)Gg*FC*4);

    const int TB = 256;
    const long NHC = (long)Nn * HC;
    const int  EHB = cdiv((long)EP * Hh, TB);
    const int  NHB = cdiv(NHC, TB);
    const unsigned int NEG_INF_BITS = 0xFF800000u;

    // weight transposes (f32 -> f16, [K,N] -> [N,K])
    transpose_f16_kernel<<<cdiv(HC*HC, TB), TB, 0, stream>>>(W2, W2t, HC, HC);
    transpose_f16_kernel<<<cdiv(HC*FC, TB), TB, 0, stream>>>(fc1_w, fc1t, HC, FC);

    // ---------------- layer 1 ----------------
    lin1_kernel<<<NHB, TB, 0, stream>>>(x, W1, P0);
    attcoef_kernel<<<cdiv(Nn*Hh, TB), TB, 0, stream>>>(P0, att_s1, att_d1, a_s, a_d);
    fill_u32_kernel<<<cdiv(Nn*Hh, TB), TB, 0, stream>>>((unsigned int*)mmax, NEG_INF_BITS, Nn*Hh);
    fill_u32_kernel<<<cdiv(Nn*Hh, TB), TB, 0, stream>>>((unsigned int*)denom, 0u, Nn*Hh);
    edge_max_kernel<<<EHB, TB, 0, stream>>>(ei, a_s, a_d, mmax);
    edge_exp_kernel<<<EHB, TB, 0, stream>>>(ei, a_s, a_d, mmax, exb, denom);
    fill_u32_kernel<<<NHB, TB, 0, stream>>>((unsigned int*)P1, 0u, NHC);
    scatter_kernel<<<cdiv((long)EP*32, TB), TB, 0, stream>>>(ei, P0, exb, denom, P1);
    bias_relu_kernel<<<NHB, TB, 0, stream>>>(P1, b1, F0);

    // ---------------- layer 2 linear: P0 = relu(h1)@W2  (WMMA, 16x64 strips) ----------------
    {
        long strips = (long)(Nn/16) * (HC/64);
        wmma_gemm_kernel<<<cdiv(strips, 8), 256, 0, stream>>>(F0, W2t, P0, nullptr, Nn, HC, HC, 0);
    }
    attcoef_kernel<<<cdiv(Nn*Hh, TB), TB, 0, stream>>>(P0, att_s2, att_d2, a_s, a_d);
    fill_u32_kernel<<<cdiv(Nn*Hh, TB), TB, 0, stream>>>((unsigned int*)mmax, NEG_INF_BITS, Nn*Hh);
    fill_u32_kernel<<<cdiv(Nn*Hh, TB), TB, 0, stream>>>((unsigned int*)denom, 0u, Nn*Hh);
    edge_max_kernel<<<EHB, TB, 0, stream>>>(ei, a_s, a_d, mmax);
    edge_exp_kernel<<<EHB, TB, 0, stream>>>(ei, a_s, a_d, mmax, exb, denom);
    fill_u32_kernel<<<NHB, TB, 0, stream>>>((unsigned int*)P1, 0u, NHC);
    scatter_kernel<<<cdiv((long)EP*32, TB), TB, 0, stream>>>(ei, P0, exb, denom, P1);
    bias_relu_kernel<<<NHB, TB, 0, stream>>>(P1, b2, nullptr);

    // ---------------- global mean pool ----------------
    fill_u32_kernel<<<cdiv(Gg*HC, TB), TB, 0, stream>>>((unsigned int*)embp, 0u, Gg*HC);
    fill_u32_kernel<<<cdiv(Gg, TB), TB, 0, stream>>>((unsigned int*)cnt, 0u, Gg);
    pool_sum_kernel<<<NHB, TB, 0, stream>>>(P1, batch, embp, cnt);
    pool_div_kernel<<<cdiv(Gg*HC, TB), TB, 0, stream>>>(embp, cnt, emb16);

    // ---------------- fc1 (WMMA, bias+relu fused) ----------------
    {
        long strips = (long)(Gg/16) * (FC/64);
        wmma_gemm_kernel<<<cdiv(strips, 8), 256, 0, stream>>>(emb16, fc1t, z, fc1_b, Gg, FC, HC, 1);
    }

    // ---------------- heads ----------------
    heads_kernel<<<cdiv(Gg, TB), TB, 0, stream>>>(z, pol_w, pol_b, val_w, val_b, out);
}
